// LSTM_25091198943705
// MI455X (gfx1250) — matrix operand order repaired and checked by reading, compiled, not run on hardware
//
#include <hip/hip_runtime.h>
#include <hip/hip_bf16.h>

// ---------------------------------------------------------------------------
// CDNA5 (gfx1250) ConvLSTM stack, wave32 WMMA.
// conv as GEMM: one wave computes a 16(M spatial) x 64(N out-channel) tile,
// 4x V_WMMA_F32_16X16X32_F16 per 32-wide K chunk, fp16 operands, fp32 accum.
// K loop is structured per conv tap (dh,dw) x Cin-chunks so the A gather is
// contiguous-channel loads with no div/mod (all Cin are multiples of 32
// except layer 1's input conv, which takes a generic fallback path once).
// Activations are carried in fp16; z / cell state stay fp32.
// ---------------------------------------------------------------------------

typedef _Float16 v16h __attribute__((ext_vector_type(16)));
typedef float    v8f  __attribute__((ext_vector_type(8)));

__global__ void cvt_f16_kernel(const float* __restrict__ in,
                               _Float16* __restrict__ out, int n) {
  int i = blockIdx.x * blockDim.x + threadIdx.x;
  if (i < n) out[i] = (_Float16)in[i];
}

// out[img, m, n] = sum_k A[m,k] * W[k,n] (+ bias[n]) (+ addin[img, m, n])
// x: fp16 NHWC, w: fp16 K-major (k*Cout+n), out/addin: fp32.
__global__ void conv_wmma(const _Float16* __restrict__ x, long x_stride,
                          const _Float16* __restrict__ w,
                          const float* __restrict__ bias,
                          const float* __restrict__ addin, long add_stride,
                          float* __restrict__ out, long out_stride,
                          int Hin, int Win, int Cin, int KH, int KW, int Cout,
                          int Ho, int Wo, int pad_h, int pad_w) {
  const int lane = threadIdx.x & 31;
  const int half = lane >> 4;
  const int mlan = lane & 15;
  const int img  = blockIdx.x;
  const int m0   = blockIdx.y * 16;
  const int n0   = blockIdx.z * 64;
  const int M    = Ho * Wo;
  const int K    = KH * KW * Cin;

  const _Float16* xi = x + (size_t)img * x_stride;

  const int  m      = m0 + mlan;
  const int  oh     = m / Wo;
  const int  ow     = m - oh * Wo;
  const bool mvalid = (m < M);

  // Per-lane K offsets within a 32-chunk (ISA 16-bit A fragment layout).
  int koff[8];
#pragma unroll
  for (int r = 0; r < 8; ++r)
    koff[r] = (r < 4) ? (half * 8 + 2 * r) : (16 + half * 8 + 2 * (r - 4));

  v8f acc[4] = {{}, {}, {}, {}};

  if ((Cin & 31) == 0) {
    // ---- Fast path: chunk = one tap x 32 contiguous channels.
    for (int dh = 0; dh < KH; ++dh) {
      const int ih = oh + dh - pad_h;
      for (int dw = 0; dw < KW; ++dw) {
        const int  iw = ow + dw - pad_w;
        const bool rv =
            mvalid && ih >= 0 && ih < Hin && iw >= 0 && iw < Win;
        const _Float16* xrow = xi + ((size_t)ih * Win + iw) * Cin;
        const int       tapk = (dh * KW + dw) * Cin;
        for (int cb = 0; cb < Cin; cb += 32) {
          v16h a;
#pragma unroll
          for (int r = 0; r < 8; ++r) {
            const int ci = cb + koff[r];
            _Float16 a0 = (_Float16)0.0f, a1 = (_Float16)0.0f;
            if (rv) { a0 = xrow[ci]; a1 = xrow[ci + 1]; }
            a[2 * r]     = a0;
            a[2 * r + 1] = a1;
          }
          const int       krow = tapk + cb;
          const _Float16* wp   = w + (size_t)krow * Cout + n0 + mlan;
          __builtin_prefetch((const void*)(wp + (size_t)32 * Cout), 0, 1);
#pragma unroll
          for (int tn = 0; tn < 4; ++tn) {
            v16h b;
#pragma unroll
            for (int n = 0; n < 16; ++n) b[n] = wp[(size_t)((n < 8) ? (half * 8 + n) : (16 + half * 8 + n - 8)) * Cout + tn * 16];
            acc[tn] = __builtin_amdgcn_wmma_f32_16x16x32_f16(
                false, a, false, b, (short)0, acc[tn], false, false);
          }
        }
      }
    }
  } else {
    // ---- Generic fallback (layer 1 input conv: Cin=1, K=6 -> one chunk).
    const int KWC = KW * Cin;
    for (int kb = 0; kb < K; kb += 32) {
      v16h a;
#pragma unroll
      for (int r = 0; r < 8; ++r) {
#pragma unroll
        for (int j = 0; j < 2; ++j) {
          const int k = kb + koff[r] + j;
          _Float16  v = (_Float16)0.0f;
          if (mvalid && k < K) {
            const int dh  = k / KWC;
            const int rem = k - dh * KWC;
            const int dw  = rem / Cin;
            const int ci  = rem - dw * Cin;
            const int ih  = oh + dh - pad_h;
            const int iw  = ow + dw - pad_w;
            if (ih >= 0 && ih < Hin && iw >= 0 && iw < Win)
              v = xi[((size_t)ih * Win + iw) * Cin + ci];
          }
          a[2 * r + j] = v;
        }
      }
      const int krow = kb;
#pragma unroll
      for (int tn = 0; tn < 4; ++tn) {
        v16h b;
#pragma unroll
        for (int n = 0; n < 16; ++n) {
          const int kk = krow + ((n < 8) ? (half * 8 + n) : (16 + half * 8 + n - 8));
          b[n] = (kk < K) ? w[(size_t)kk * Cout + n0 + tn * 16 + mlan] : (_Float16)0.0f;
        }
        acc[tn] = __builtin_amdgcn_wmma_f32_16x16x32_f16(
            false, a, false, b, (short)0, acc[tn], false, false);
      }
    }
  }

  // ---- Store C/D: VGPR r holds M = r + 8*half, N = lane&15 per 16-tile.
  const float* ai = addin ? (addin + (size_t)img * add_stride) : nullptr;
  float*       oi = out + (size_t)img * out_stride;
#pragma unroll
  for (int tn = 0; tn < 4; ++tn) {
    const int   n    = n0 + tn * 16 + mlan;
    const float badd = bias ? bias[n] : 0.0f;
#pragma unroll
    for (int r = 0; r < 8; ++r) {
      const int mr = m0 + r + 8 * half;
      if (mr < M) {
        float v = acc[tn][r] + badd;
        if (ai) v += ai[(size_t)mr * Cout + n];
        oi[(size_t)mr * Cout + n] = v;
      }
    }
  }
}

// Elementwise LSTM gate update; h written as fp16 (next layer's A operand).
__global__ void lstm_gate(const float* __restrict__ z, float* __restrict__ c,
                          _Float16* __restrict__ h, _Float16* __restrict__ hseq,
                          long hseq_bstride, int npos, int f, int B) {
  long idx   = (long)blockIdx.x * blockDim.x + threadIdx.x;
  long total = (long)B * npos * f;
  if (idx >= total) return;
  const int  j  = (int)(idx % f);
  const long pb = idx / f;  // b*npos + p
  const int  b  = (int)(pb / npos);

  const long  zbase = pb * (4L * f);
  const float zi = z[zbase + j];
  const float zf = z[zbase + f + j];
  const float zc = z[zbase + 2L * f + j];
  const float zo = z[zbase + 3L * f + j];

  auto hs = [](float v) { v = 0.2f * v + 0.5f; return fminf(fmaxf(v, 0.f), 1.f); };
  const float cc = hs(zf) * c[idx] + hs(zi) * fmaxf(zc, 0.f);
  const float hh = hs(zo) * fmaxf(cc, 0.f);
  c[idx] = cc;
  h[idx] = (_Float16)hh;
  if (hseq) {
    const long p = pb - (long)b * npos;
    hseq[(long)b * hseq_bstride + p * (long)f + j] = (_Float16)hh;
  }
}

// 2x2 max pool (floor) on fp16 NHWC images.
__global__ void maxpool2(const _Float16* __restrict__ in,
                         _Float16* __restrict__ out, int nimg, int H, int W,
                         int C) {
  const int Hp = H / 2, Wp = W / 2;
  long idx   = (long)blockIdx.x * blockDim.x + threadIdx.x;
  long total = (long)nimg * Hp * Wp * C;
  if (idx >= total) return;
  const int  cc = (int)(idx % C);
  long       t  = idx / C;
  const int  wp = (int)(t % Wp); t /= Wp;
  const int  hp = (int)(t % Hp);
  const int  im = (int)(t / Hp);
  const long b00 = (((long)im * H + 2 * hp) * W + 2 * wp) * C + cc;
  const long rs  = (long)W * C;
  float v = (float)in[b00];
  v = fmaxf(v, (float)in[b00 + C]);
  v = fmaxf(v, (float)in[b00 + rs]);
  v = fmaxf(v, (float)in[b00 + rs + C]);
  out[idx] = (_Float16)v;
}

// Dense head: out[b] = (h[b,:] @ W1 + b1) @ W2 + b2. One wave per batch row.
__global__ void dense_head(const _Float16* __restrict__ hin,
                           const float* __restrict__ w1,
                           const float* __restrict__ b1,
                           const float* __restrict__ w2,
                           const float* __restrict__ b2,
                           float* __restrict__ out, int Kd, int N1) {
  const int b = blockIdx.x;
  const int j = threadIdx.x;
  __shared__ float tmp[32];
  float s = 0.f;
  if (j < N1) {
    const _Float16* hb = hin + (long)b * Kd;
    for (int k = 0; k < Kd; ++k) s += (float)hb[k] * w1[k * N1 + j];
    s += b1[j];
  }
  tmp[j] = s;
  __syncthreads();
  if (j == 0) {
    float o = b2[0];
    for (int q = 0; q < N1; ++q) o += tmp[q] * w2[q];
    out[b] = o;
  }
}

// ---------------------------------------------------------------------------
// Host orchestration
// ---------------------------------------------------------------------------
static void convlstm_layer(hipStream_t stream, const _Float16* xin,
                           const float* bias, const _Float16* wx16,
                           const _Float16* wh16, int B, int T, int Hin, int Win,
                           int Cin, int KH, int KW, int f, float* zin,
                           float* zt, float* cbuf, _Float16* hbuf,
                           _Float16* hseq, bool return_seq, int& Ho, int& Wo) {
  Ho = Hin - KH + 1;
  Wo = Win - KW + 1;
  const int    Cout = 4 * f;
  const size_t ipos = (size_t)Ho * Wo;

  // Input conv for all B*T images at once (VALID).
  dim3 g1(B * T, (unsigned)((ipos + 15) / 16), (unsigned)(Cout / 64));
  conv_wmma<<<g1, 32, 0, stream>>>(xin, (long)Hin * Win * Cin, wx16, bias,
                                   nullptr, 0, zin, (long)(ipos * Cout), Hin,
                                   Win, Cin, KH, KW, Cout, Ho, Wo, 0, 0);

  hipMemsetAsync(cbuf, 0, (size_t)B * ipos * f * sizeof(float), stream);
  hipMemsetAsync(hbuf, 0, (size_t)B * ipos * f * sizeof(_Float16), stream);

  const int pad_h = (KH - 1) / 2;  // lax 'SAME' pad_lo; high side via bounds
  const int pad_w = (KW - 1) / 2;
  for (int t = 0; t < T; ++t) {
    dim3 g2(B, (unsigned)((ipos + 15) / 16), (unsigned)(Cout / 64));
    conv_wmma<<<g2, 32, 0, stream>>>(
        hbuf, (long)(ipos * f), wh16, nullptr, zin + (size_t)t * ipos * Cout,
        (long)((size_t)T * ipos * Cout), zt, (long)(ipos * Cout), Ho, Wo, f,
        KH, KW, Cout, Ho, Wo, pad_h, pad_w);

    const long total = (long)B * ipos * f;
    _Float16* hs = return_seq ? (hseq + (size_t)t * ipos * f) : nullptr;
    lstm_gate<<<(unsigned)((total + 255) / 256), 256, 0, stream>>>(
        zt, cbuf, hbuf, hs, (long)((size_t)T * ipos * f), (int)ipos, f, B);
  }
}

extern "C" void kernel_launch(void* const* d_in, const int* in_sizes, int n_in,
                              void* d_out, int out_size, void* d_ws,
                              size_t ws_size, hipStream_t stream) {
  const int B = 16, T = 10;
  const float* x = (const float*)d_in[0];
  const float* wx[6];
  const float* wh[6];
  const float* bias[6];
  for (int i = 0; i < 6; ++i) {
    wx[i]   = (const float*)d_in[1 + 3 * i];
    wh[i]   = (const float*)d_in[2 + 3 * i];
    bias[i] = (const float*)d_in[3 + 3 * i];
  }
  const float* d1w = (const float*)d_in[19];
  const float* d1b = (const float*)d_in[20];
  const float* d2w = (const float*)d_in[21];
  const float* d2b = (const float*)d_in[22];

  // Bump allocator over d_ws.
  char* p = (char*)d_ws;
  auto  alloc = [&](size_t bytes) -> void* {
    void* r = p;
    p += (bytes + 255) & ~(size_t)255;
    return r;
  };

  // fp32 -> fp16 weight conversion (memory layout is already K-major).
  _Float16* wx16[6];
  _Float16* wh16[6];
  for (int i = 0; i < 6; ++i) {
    const int szx = in_sizes[1 + 3 * i];
    const int szh = in_sizes[2 + 3 * i];
    wx16[i] = (_Float16*)alloc((size_t)szx * sizeof(_Float16));
    wh16[i] = (_Float16*)alloc((size_t)szh * sizeof(_Float16));
    cvt_f16_kernel<<<(szx + 255) / 256, 256, 0, stream>>>(wx[i], wx16[i], szx);
    cvt_f16_kernel<<<(szh + 255) / 256, 256, 0, stream>>>(wh[i], wh16[i], szh);
  }
  // Input x -> fp16.
  const int xsz = in_sizes[0];  // B*T*32*64*1
  _Float16* x16 = (_Float16*)alloc((size_t)xsz * sizeof(_Float16));
  cvt_f16_kernel<<<(xsz + 255) / 256, 256, 0, stream>>>(x, x16, xsz);

  const size_t F = sizeof(float), H2 = sizeof(_Float16);
  // Max per-buffer footprints across all layers (layer 1 dominates).
  float*    zin  = (float*)alloc((size_t)B * T * 31 * 62 * 128 * F);
  float*    zt   = (float*)alloc((size_t)B * 31 * 62 * 128 * F);
  float*    cbuf = (float*)alloc((size_t)B * 31 * 62 * 32 * F);
  _Float16* hbuf = (_Float16*)alloc((size_t)B * 31 * 62 * 32 * H2);
  _Float16* seqA = (_Float16*)alloc((size_t)B * T * 31 * 62 * 32 * H2);
  _Float16* seqB = (_Float16*)alloc((size_t)B * T * 30 * 60 * 32 * H2);

  int Ho, Wo;
  // L1: (32,64,1) k(2,3) f=32 -> seqA (31,62,32)
  convlstm_layer(stream, x16, bias[0], wx16[0], wh16[0], B, T, 32, 64, 1, 2, 3,
                 32, zin, zt, cbuf, hbuf, seqA, true, Ho, Wo);
  // L2: (31,62,32) -> seqB (30,60,32)
  convlstm_layer(stream, seqA, bias[1], wx16[1], wh16[1], B, T, 31, 62, 32, 2,
                 3, 32, zin, zt, cbuf, hbuf, seqB, true, Ho, Wo);
  // pool1: (30,60,32) -> seqA (15,30,32)
  {
    long tot = (long)B * T * 15 * 30 * 32;
    maxpool2<<<(unsigned)((tot + 255) / 256), 256, 0, stream>>>(
        seqB, seqA, B * T, 30, 60, 32);
  }
  // L3: (15,30,32) f=64 -> seqB (14,28,64)
  convlstm_layer(stream, seqA, bias[2], wx16[2], wh16[2], B, T, 15, 30, 32, 2,
                 3, 64, zin, zt, cbuf, hbuf, seqB, true, Ho, Wo);
  // L4: (14,28,64) -> seqA (13,26,64)
  convlstm_layer(stream, seqB, bias[3], wx16[3], wh16[3], B, T, 14, 28, 64, 2,
                 3, 64, zin, zt, cbuf, hbuf, seqA, true, Ho, Wo);
  // pool2: (13,26,64) -> seqB (6,13,64)
  {
    long tot = (long)B * T * 6 * 13 * 64;
    maxpool2<<<(unsigned)((tot + 255) / 256), 256, 0, stream>>>(
        seqA, seqB, B * T, 13, 26, 64);
  }
  // L5: (6,13,64) k(1,3) f=128 -> seqA (6,11,128)
  convlstm_layer(stream, seqB, bias[4], wx16[4], wh16[4], B, T, 6, 13, 64, 1,
                 3, 128, zin, zt, cbuf, hbuf, seqA, true, Ho, Wo);
  // L6: (6,11,128) k(1,3) f=128, return_sequences=False -> final h in hbuf
  convlstm_layer(stream, seqA, bias[5], wx16[5], wh16[5], B, T, 6, 11, 128, 1,
                 3, 128, zin, zt, cbuf, hbuf, nullptr, false, Ho, Wo);
  // pool3: hbuf (B,6,9,128) -> seqB (B,3,4,128) == (B,1536)
  {
    long tot = (long)B * 3 * 4 * 128;
    maxpool2<<<(unsigned)((tot + 255) / 256), 256, 0, stream>>>(hbuf, seqB, B,
                                                               6, 9, 128);
  }
  // Dense head: 1536 -> 10 -> 1
  dense_head<<<B, 32, 0, stream>>>(seqB, d1w, d1b, d2w, d2b, (float*)d_out,
                                   1536, 10);
  (void)n_in; (void)out_size; (void)ws_size;
}
